// Model_2808908611975
// MI455X (gfx1250) — matrix operand order repaired and checked by reading, not compiled, not run on hardware
//
#include <hip/hip_runtime.h>
#include <hip/hip_bf16.h>
#include <math.h>

typedef __attribute__((ext_vector_type(16))) _Float16 v16h;
typedef __attribute__((ext_vector_type(8)))  float    v8f;

#define N_NODES 8192
#define IN_DIM  256
#define H1D     128
#define H2D     64

// pack two fp32 -> packed f16 pair (v_cvt_pk_rtz_f16_f32)
static __device__ __forceinline__ unsigned pk2(float a, float b) {
    auto h = __builtin_amdgcn_cvt_pkrtz(a, b);
    return __builtin_bit_cast(unsigned, h);
}

// ---------------------------------------------------------------------------
// Tiled GEMM for the two small fp32-input layers:
//   C[M,128] = A[M,K] @ B[K,128], fp32 in/out, f16 WMMA, fp32 accumulate.
// Block: 256 threads = 8 waves; block tile 16 (M) x 128 (N); one 16x16 WMMA
// tile per wave; K staged in chunks of 32 through LDS with fp32->f16 convert.
// ---------------------------------------------------------------------------
__global__ __launch_bounds__(256)
void gemm_f16_wmma(const float* __restrict__ A, const float* __restrict__ B,
                   float* __restrict__ C, int M, int Ncols, int K)
{
    __shared__ unsigned As2[16 * 17];    // 16 rows x 16 K-pairs (+1 pad)
    __shared__ unsigned Bs2[16 * 132];   // 16 K-pairs x 128 cols (+4 pad)

    const int t    = threadIdx.x;
    const int wave = t >> 5;
    const int lane = t & 31;
    const int m16  = lane & 15;
    const int kh   = lane >> 4;          // lane half selects K sub-range

    const int row0 = blockIdx.x * 16;

    v8f acc = {};

    for (int k0 = 0; k0 < K; k0 += 32) {
        __syncthreads();
        // ---- stage A tile: 16 x 32 fp32 -> f16 pairs (2 elements / thread)
        {
            int idx = t * 2;
            int r   = idx >> 5;
            int k   = idx & 31;
            const float* ap = A + (size_t)(row0 + r) * K + k0 + k;
            float2 f = *(const float2*)ap;
            As2[r * 17 + (k >> 1)] = pk2(f.x, f.y);
        }
        // ---- stage B tile: 32 x 128, stored as K-pair-major [16][132]
        {
            int k2 = t >> 4;              // K pair index 0..15
            int n  = (t & 15) * 8;        // 8 consecutive columns
            const float* r0p = B + (size_t)(k0 + 2 * k2) * Ncols + n;
            const float* r1p = r0p + Ncols;
#pragma unroll
            for (int j = 0; j < 8; ++j)
                Bs2[k2 * 132 + n + j] = pk2(r0p[j], r1p[j]);
        }
        __syncthreads();

        // ---- build WMMA fragments per ISA 7.12.2 layouts
        union { v16h v; unsigned u[8]; } af, bf;
#pragma unroll
        for (int i = 0; i < 8; ++i)       // A: K pair = (i>>2)*8 + kh*4 + (i&3)
            af.u[i] = As2[m16 * 17 + ((i >> 2) * 8 + kh * 4 + (i & 3))];
        const int bcol = wave * 16 + m16;
#pragma unroll
        for (int j = 0; j < 8; ++j)       // B: K pair = (j>>2)*8 + kh*4 + (j&3)
            bf.u[j] = Bs2[((j >> 2) * 8 + kh * 4 + (j & 3)) * 132 + bcol];

        acc = __builtin_amdgcn_wmma_f32_16x16x32_f16(
                  false, af.v, false, bf.v, (short)0, acc, false, false);
    }

    const int col = wave * 16 + m16;
#pragma unroll
    for (int r = 0; r < 8; ++r)           // C: row = r + 8*kh, col = lane&15
        C[(size_t)(row0 + r + kh * 8) * Ncols + col] = acc[r];
}

// ---------------------------------------------------------------------------
// Dominant kernel: out = nan_to_num(sigmoid(z @ z^T)), z16 f16 [8192 x 64].
// No LDS: fragments are contiguous b128 loads from the 1 MB L2-resident z16.
// Block: 256 threads = 8 waves; block tile 16 (M) x 512 (N); each wave owns
// 4 column tiles (16x64) -> 8 WMMAs per wave.  The 256 MB output exceeds the
// 192 MB L2, so the epilogue uses non-temporal stores (TH=NT) to stream to
// HBM without evicting the z16 operand working set.
// ---------------------------------------------------------------------------
__global__ __launch_bounds__(256)
void zzt_sigmoid_wmma(const unsigned short* __restrict__ z16,
                      float* __restrict__ out)
{
    const int t    = threadIdx.x;
    const int wave = t >> 5;
    const int lane = t & 31;
    const int m16  = lane & 15;
    const int kh   = lane >> 4;

    const int row0 = blockIdx.x * 16;
    const int col0 = blockIdx.y * 512 + wave * 64;

    union F { v16h v; uint4 q[2]; };

    // A fragments for K=0..31 and K=32..63 (row = row0 + m16)
    const uint4* arow = (const uint4*)(z16 + (size_t)(row0 + m16) * H2D);
    F a0, a1;
    a0.q[0] = arow[0 + kh];   // halves [kh*8,      +8)
    a0.q[1] = arow[2 + kh];   // halves [16+kh*8,   +8)
    a1.q[0] = arow[4 + kh];   // halves [32+kh*8,   +8)
    a1.q[1] = arow[6 + kh];   // halves [48+kh*8,   +8)

#pragma unroll
    for (int c = 0; c < 4; ++c) {
        const int col = col0 + c * 16 + m16;
        const uint4* brow = (const uint4*)(z16 + (size_t)col * H2D);
        F b0, b1;
        b0.q[0] = brow[0 + kh];   // halves [kh*8,      +8)
        b0.q[1] = brow[2 + kh];   // halves [16+kh*8,   +8)
        b1.q[0] = brow[4 + kh];   // halves [32+kh*8,   +8)
        b1.q[1] = brow[6 + kh];   // halves [48+kh*8,   +8)

        v8f acc = {};
        acc = __builtin_amdgcn_wmma_f32_16x16x32_f16(
                  false, a0.v, false, b0.v, (short)0, acc, false, false);
        acc = __builtin_amdgcn_wmma_f32_16x16x32_f16(
                  false, a1.v, false, b1.v, (short)0, acc, false, false);

#pragma unroll
        for (int r = 0; r < 8; ++r) {     // C: row = r + 8*kh, col = lane&15
            float v = acc[r];
            v = 1.0f / (1.0f + expf(-v));
            if (v != v) v = 0.0f;         // nan_to_num
            // streaming store: TH=NT, bypass L2 residency
            __builtin_nontemporal_store(
                v, &out[(size_t)(row0 + r + kh * 8) * N_NODES + col]);
        }
    }
}

// ---------------------------------------------------------------------------
// Degree / normalization
// ---------------------------------------------------------------------------
__global__ void k_init_deg(float* deg) {
    int i = blockIdx.x * blockDim.x + threadIdx.x;
    if (i < N_NODES) deg[i] = 1.0f;                       // self loop
}

__global__ void k_count_deg(const long long* __restrict__ dst, float* deg, int E) {
    int e = blockIdx.x * blockDim.x + threadIdx.x;
    if (e < E) atomicAdd(&deg[(int)dst[e]], 1.0f);
}

__global__ void k_dinv(const float* __restrict__ deg, float* __restrict__ dinv) {
    int i = blockIdx.x * blockDim.x + threadIdx.x;
    if (i < N_NODES) dinv[i] = rsqrtf(fmaxf(deg[i], 1.0f));
}

// ---------------------------------------------------------------------------
// GCN aggregation: out[i] = dinv[i]^2 * feat[i] + bias  (self-loop + bias),
// then edge atomics add dinv[s]*dinv[d] * feat[s] into out[d].  F == 128.
// ---------------------------------------------------------------------------
__global__ void k_scatter_init(const float* __restrict__ feat,
                               const float* __restrict__ dinv,
                               const float* __restrict__ biasA,
                               const float* __restrict__ biasB, int split,
                               float* __restrict__ out)
{
    int tid = blockIdx.x * blockDim.x + threadIdx.x;      // N*128 threads
    int i = tid >> 7, f = tid & 127;
    float b = (f < split) ? biasA[f] : biasB[f - split];
    float w = dinv[i] * dinv[i];
    out[tid] = w * feat[tid] + b;
}

__global__ void k_scatter_edges(const long long* __restrict__ srcI,
                                const long long* __restrict__ dstI,
                                const float* __restrict__ feat,
                                const float* __restrict__ dinv,
                                float* __restrict__ out, int E)
{
    int tid = blockIdx.x * blockDim.x + threadIdx.x;      // E*32 threads
    int e = tid >> 5;
    if (e >= E) return;
    int c = (tid & 31) * 4;
    int s = (int)srcI[e], d = (int)dstI[e];
    float w = dinv[s] * dinv[d];
    const float* fp = feat + (size_t)s * 128 + c;
    float*       op = out  + (size_t)d * 128 + c;
    atomicAdd(op + 0, w * fp[0]);
    atomicAdd(op + 1, w * fp[1]);
    atomicAdd(op + 2, w * fp[2]);
    atomicAdd(op + 3, w * fp[3]);
}

// pack [Wm | Ws] -> Wcat[128][128] so both output heads run in one GEMM
__global__ void k_pack_w(const float* __restrict__ Wm, const float* __restrict__ Ws,
                         float* __restrict__ Wcat)
{
    int tid = blockIdx.x * blockDim.x + threadIdx.x;      // 128*128
    int k = tid >> 7, n = tid & 127;
    Wcat[tid] = (n < 64) ? Wm[k * 64 + n] : Ws[k * 64 + (n - 64)];
}

// z = mean + noise * exp(log_std)  ->  f16 (feeds WMMA directly)
__global__ void k_reparam_f16(const float* __restrict__ ml,
                              const float* __restrict__ noise,
                              unsigned short* __restrict__ z16)
{
    int tid = blockIdx.x * blockDim.x + threadIdx.x;      // N*64
    int i = tid >> 6, j = tid & 63;
    float mean = ml[(size_t)i * 128 + j];
    float ls   = ml[(size_t)i * 128 + 64 + j];
    float zv   = mean + noise[tid] * expf(ls);
    _Float16 h = (_Float16)zv;
    z16[tid] = __builtin_bit_cast(unsigned short, h);
}

// ---------------------------------------------------------------------------
extern "C" void kernel_launch(void* const* d_in, const int* in_sizes, int n_in,
                              void* d_out, int out_size, void* d_ws, size_t ws_size,
                              hipStream_t stream)
{
    const float*     X     = (const float*)d_in[0];
    const long long* ei    = (const long long*)d_in[1];   // int64 [2,E]
    const float*     noise = (const float*)d_in[2];
    const float*     W0    = (const float*)d_in[3];
    const float*     b0    = (const float*)d_in[4];
    const float*     Wm    = (const float*)d_in[5];
    const float*     bm    = (const float*)d_in[6];
    const float*     Ws    = (const float*)d_in[7];
    const float*     bs    = (const float*)d_in[8];
    float* out = (float*)d_out;

    const int E = in_sizes[1] / 2;
    const long long* src = ei;
    const long long* dst = ei + E;

    // workspace carve-out (~18 MB)
    char* ws = (char*)d_ws;
    size_t off = 0;
    auto carve = [&](size_t bytes) -> char* {
        char* p = ws + off;
        off = (off + bytes + 255) & ~(size_t)255;
        return p;
    };
    float* deg  = (float*)carve((size_t)N_NODES * 4);
    float* dinv = (float*)carve((size_t)N_NODES * 4);
    float* hw   = (float*)carve((size_t)N_NODES * 128 * 4);   // X @ W0
    float* h    = (float*)carve((size_t)N_NODES * 128 * 4);   // aggregated layer 1
    float* Wcat = (float*)carve((size_t)128 * 128 * 4);       // [Wm | Ws]
    float* msw  = (float*)carve((size_t)N_NODES * 128 * 4);   // h @ Wcat
    float* ml   = (float*)carve((size_t)N_NODES * 128 * 4);   // aggregated [mean|log_std]
    unsigned short* z16 = (unsigned short*)carve((size_t)N_NODES * 64 * 2);

    // 1) symmetric normalization
    k_init_deg <<<N_NODES / 256, 256, 0, stream>>>(deg);
    k_count_deg<<<(E + 255) / 256, 256, 0, stream>>>(dst, deg, E);
    k_dinv     <<<N_NODES / 256, 256, 0, stream>>>(deg, dinv);

    // 2) layer 1: hw = X @ W0  (8192x256 @ 256x128)
    gemm_f16_wmma<<<dim3(N_NODES / 16, 1), 256, 0, stream>>>(
        X, W0, hw, N_NODES, H1D, IN_DIM);

    //    aggregate: h[d] += norm * hw[s], + b0
    k_scatter_init <<<(N_NODES * 128) / 256, 256, 0, stream>>>(hw, dinv, b0, b0, 128, h);
    k_scatter_edges<<<((size_t)E * 32 + 255) / 256, 256, 0, stream>>>(src, dst, hw, dinv, h, E);

    // 3) fused mean/log_std: msw = h @ [Wm|Ws]  (8192x128 @ 128x128)
    k_pack_w<<<(128 * 128) / 256, 256, 0, stream>>>(Wm, Ws, Wcat);
    gemm_f16_wmma<<<dim3(N_NODES / 16, 1), 256, 0, stream>>>(
        h, Wcat, msw, N_NODES, 128, H1D);

    //    aggregate with bias [bm|bs]
    k_scatter_init <<<(N_NODES * 128) / 256, 256, 0, stream>>>(msw, dinv, bm, bs, 64, ml);
    k_scatter_edges<<<((size_t)E * 32 + 255) / 256, 256, 0, stream>>>(src, dst, msw, dinv, ml, E);

    // 4) reparameterize -> f16 z (1 MB, L2-resident)
    k_reparam_f16<<<(N_NODES * 64) / 256, 256, 0, stream>>>(ml, noise, z16);

    // 5) out = nan_to_num(sigmoid(z @ z^T)) -> 256 MB NT store (HBM roofline)
    zzt_sigmoid_wmma<<<dim3(N_NODES / 16, N_NODES / 512), 256, 0, stream>>>(z16, out);
}
